// BinLinear_12635793784935
// MI455X (gfx1250) — matrix operand order repaired and checked by reading, not compiled, not run on hardware
//
#include <hip/hip_runtime.h>
#include <stdint.h>

#define N_ROWS 8192
#define K_DIM  4096
#define N_COLS 4096

#define BM 256
#define BN 128
#define BK 32
#define NKT (K_DIM / BK)      // 128 k-steps
#define LDS_ROW 80            // 32 bf16 = 64B data + 16B pad (bank-conflict-free, 16B aligned)

typedef __attribute__((ext_vector_type(16))) __bf16 v16bf;
typedef __attribute__((ext_vector_type(8)))  __bf16 v8bf;
typedef __attribute__((ext_vector_type(8)))  float  v8f;

__device__ __forceinline__ uint16_t f32_to_bf16_rne(float f) {
    union { float f; uint32_t u; } v; v.f = f;
    uint32_t r = v.u + 0x7FFFu + ((v.u >> 16) & 1u);
    return (uint16_t)(r >> 16);
}

// ---- preprocessing: X fp32 -> bf16 -------------------------------------
__global__ void cvt_x_kernel(const float* __restrict__ x,
                             uint16_t* __restrict__ xb, long n) {
    long tid = (long)blockIdx.x * blockDim.x + threadIdx.x;
    long stride = (long)gridDim.x * blockDim.x;
    for (long i = tid * 4; i < n; i += stride * 4) {
        float4 v = *(const float4*)(x + i);
        ushort4 o;
        o.x = f32_to_bf16_rne(v.x);
        o.y = f32_to_bf16_rne(v.y);
        o.z = f32_to_bf16_rne(v.z);
        o.w = f32_to_bf16_rne(v.w);
        *(ushort4*)(xb + i) = o;
    }
}

// ---- preprocessing: W fp32 -> bf16 sign(clip(w,-1,1)) in {-1,0,+1} -----
__device__ __forceinline__ uint16_t sign_bf16(float w) {
    return (w > 0.0f) ? 0x3F80u : ((w < 0.0f) ? 0xBF80u : 0u);
}

__global__ void cvt_w_kernel(const float* __restrict__ w,
                             uint16_t* __restrict__ wb, long n) {
    long tid = (long)blockIdx.x * blockDim.x + threadIdx.x;
    long stride = (long)gridDim.x * blockDim.x;
    for (long i = tid * 4; i < n; i += stride * 4) {
        float4 v = *(const float4*)(w + i);
        ushort4 o;
        o.x = sign_bf16(v.x);
        o.y = sign_bf16(v.y);
        o.z = sign_bf16(v.z);
        o.w = sign_bf16(v.w);
        *(ushort4*)(wb + i) = o;
    }
}

// ---- CDNA5 async global -> LDS copy (16B per lane, ASYNCcnt tracked) ---
__device__ __forceinline__ void async_cp16(uint32_t lds_addr, const void* gaddr) {
    asm volatile("global_load_async_to_lds_b128 %0, %1, off"
                 :: "v"(lds_addr), "v"(gaddr) : "memory");
}

// ---- GEMM: out(8192x4096) = Xb(8192x4096) * Wb(4096x4096)^T ------------
__global__ __launch_bounds__(256) void bingemm_kernel(
    const uint16_t* __restrict__ Xb, const uint16_t* __restrict__ Wb,
    float* __restrict__ out)
{
    __shared__ __align__(16) unsigned char lA[2][BM * LDS_ROW]; // 2 x 20 KB
    __shared__ __align__(16) unsigned char lB[2][BN * LDS_ROW]; // 2 x 10 KB

    const int tid   = threadIdx.x;
    const int lane  = tid & 31;
    const int wave  = tid >> 5;      // 0..7
    const int waveM = wave & 3;      // 4 waves along M
    const int waveN = wave >> 2;     // 2 waves along N
    const int l16   = lane >> 4;     // 0/1 half-wave
    const int lrow  = lane & 15;
    const int bm = blockIdx.y * BM;
    const int bn = blockIdx.x * BN;

    // issue double-buffered tile copy: A tile 256x32 bf16 (1024 x 16B chunks),
    // B tile 128x32 bf16 (512 x 16B chunks). 6 async ops per thread per step.
    auto issue_tile = [&](int buf, int kbase) {
#pragma unroll
        for (int t = 0; t < 4; ++t) {
            int c = tid + t * 256;           // 0..1023
            int row = c >> 2, seg = c & 3;   // 4 x 16B segments per 64B row
            uint32_t lds = (uint32_t)(size_t)&lA[buf][row * LDS_ROW + seg * 16];
            const void* g = Xb + (long)(bm + row) * K_DIM + kbase + seg * 8;
            async_cp16(lds, g);
        }
#pragma unroll
        for (int t = 0; t < 2; ++t) {
            int c = tid + t * 256;           // 0..511
            int row = c >> 2, seg = c & 3;
            uint32_t lds = (uint32_t)(size_t)&lB[buf][row * LDS_ROW + seg * 16];
            const void* g = Wb + (long)(bn + row) * K_DIM + kbase + seg * 8;
            async_cp16(lds, g);
        }
    };

    const v8f vzero = {0.f, 0.f, 0.f, 0.f, 0.f, 0.f, 0.f, 0.f};
    v8f acc[4][4];
#pragma unroll
    for (int i = 0; i < 4; ++i)
#pragma unroll
        for (int j = 0; j < 4; ++j) acc[i][j] = vzero;

    issue_tile(0, 0);

    for (int kt = 0; kt < NKT; ++kt) {
        const int cur = kt & 1;
        if (kt + 1 < NKT) {
            issue_tile(cur ^ 1, (kt + 1) * BK);
            // in-order completion: <=6 outstanding means current buffer landed
            asm volatile("s_wait_asynccnt 6" ::: "memory");
        } else {
            asm volatile("s_wait_asynccnt 0" ::: "memory");
        }
        __syncthreads();

        const unsigned char* Ab = lA[cur];
        const unsigned char* Bb = lB[cur];

        // A 16x32 bf16 fragment (ISA layout): lane<16: K 0..7 | 16..23,
        // lane>=16: K 8..15 | 24..31, row M = lane%16
        v16bf afrag[4];
#pragma unroll
        for (int i = 0; i < 4; ++i) {
            const unsigned char* p =
                Ab + (waveM * 64 + i * 16 + lrow) * LDS_ROW + l16 * 16;
            v8bf lo = *(const v8bf*)(p);
            v8bf hi = *(const v8bf*)(p + 32);
            afrag[i] = __builtin_shufflevector(lo, hi,
                0,1,2,3,4,5,6,7,8,9,10,11,12,13,14,15);
        }
        // B 32x16 bf16 fragment: column N = lane%16 (a row of W),
        // lanes 0-15 hold K 0..7 | 16..23, lanes 16-31 hold K 8..15 | 24..31
        v16bf bfrag[4];
#pragma unroll
        for (int j = 0; j < 4; ++j) {
            const unsigned char* p =
                Bb + (waveN * 64 + j * 16 + lrow) * LDS_ROW + l16 * 16;
            v8bf lo = *(const v8bf*)(p);
            v8bf hi = *(const v8bf*)(p + 32);
            bfrag[j] = __builtin_shufflevector(lo, hi,
                0,1,2,3,4,5,6,7,8,9,10,11,12,13,14,15);
        }

#pragma unroll
        for (int i = 0; i < 4; ++i)
#pragma unroll
            for (int j = 0; j < 4; ++j)
                acc[i][j] = __builtin_amdgcn_wmma_f32_16x16x32_bf16(
                    false, afrag[i], false, bfrag[j],
                    (short)0, acc[i][j], false, false);

        __syncthreads();  // protect buf[cur] before next iteration overwrites it
    }

    // C layout: VGPR v: M = v + 8*(lane/16), N = lane%16
#pragma unroll
    for (int i = 0; i < 4; ++i)
#pragma unroll
        for (int j = 0; j < 4; ++j) {
            int col = bn + waveN * 64 + j * 16 + lrow;
#pragma unroll
            for (int v = 0; v < 8; ++v) {
                int row = bm + waveM * 64 + i * 16 + v + 8 * l16;
                out[(long)row * N_COLS + col] = acc[i][j][v];
            }
        }
}

extern "C" void kernel_launch(void* const* d_in, const int* in_sizes, int n_in,
                              void* d_out, int out_size, void* d_ws, size_t ws_size,
                              hipStream_t stream) {
    const float* x = (const float*)d_in[0];
    const float* w = (const float*)d_in[1];
    float* out = (float*)d_out;

    uint16_t* xb = (uint16_t*)d_ws;                               // 64 MB
    uint16_t* wb = (uint16_t*)((char*)d_ws +
                               (size_t)N_ROWS * K_DIM * sizeof(uint16_t)); // +32 MB

    long nx = (long)N_ROWS * K_DIM;
    long nw = (long)N_COLS * K_DIM;
    cvt_x_kernel<<<4096, 256, 0, stream>>>(x, xb, nx);
    cvt_w_kernel<<<2048, 256, 0, stream>>>(w, wb, nw);

    dim3 grid(N_COLS / BN, N_ROWS / BM);  // (32, 32)
    bingemm_kernel<<<grid, 256, 0, stream>>>(xb, wb, out);
}
